// CrossAttention_57664230916185
// MI455X (gfx1250) — matrix operand order repaired and checked by reading, not compiled, not run on hardware
//
#include <hip/hip_runtime.h>
#include <stdint.h>

// ---------------------------------------------------------------------------
// CDNA5 (gfx1250, wave32) attention forward.  B=2, N=4096, D=512, H=8, Dh=64.
// All matmuls via v_wmma_f32_16x16x32_f16 (f32 accum).  Gammas + softmax scale
// folded into pre-converted f16 weights.  Flash-attention stage K/V tiles into
// LDS with double-buffered global_load_async_to_lds_b128 (ASYNCcnt).
// ---------------------------------------------------------------------------

typedef __attribute__((ext_vector_type(16))) _Float16 v16h;
typedef __attribute__((ext_vector_type(8)))  _Float16 v8h;
typedef __attribute__((ext_vector_type(8)))  float    v8f;

static __device__ __forceinline__ v8f wmma_f16(v16h a, v16h b, v8f c) {
  return __builtin_amdgcn_wmma_f32_16x16x32_f16(false, a, false, b, (short)0, c,
                                                false, false);
}

static __device__ __forceinline__ v16h cat8(v8h lo, v8h hi) {
  return __builtin_shufflevector(lo, hi, 0, 1, 2, 3, 4, 5, 6, 7, 8, 9, 10, 11,
                                 12, 13, 14, 15);
}

// A-fragment (16x32 f16): lane L (row = L%16, half = L/16):
//   elems 0..7  -> K = 8*half + j ; elems 8..15 -> K = 16 + 8*half + (j-8)
static __device__ __forceinline__ v16h load_a_h(const _Float16* __restrict__ base,
                                                long ld, long row, int k0, int half) {
  const _Float16* p = base + row * ld + k0 + 8 * half;
  v8h lo = *(const v8h*)p;
  v8h hi = *(const v8h*)(p + 16);
  return cat8(lo, hi);
}

// B-fragment (32x16 f16): lane L (col N = L%16):
//   elems 0..7  -> K = 8*half + j ; elems 8..15 -> K = 16 + 8*half + (j-8)
static __device__ __forceinline__ v16h load_b_h(const _Float16* __restrict__ base,
                                                long ld, long col, int k0, int half) {
  const _Float16* p = base + col * ld + k0 + 8 * half;
  v8h lo = *(const v8h*)p;
  v8h hi = *(const v8h*)(p + 16);
  return cat8(lo, hi);
}

// ---------------------------------------------------------------------------
// One-time conversions (memory bound, ~21 MB total).
// ---------------------------------------------------------------------------
__global__ __launch_bounds__(256) void cvt_x(const float* __restrict__ x,
                                             _Float16* __restrict__ Xh) {
  long i = ((long)blockIdx.x * 256 + threadIdx.x) * 8;
  float4 a = *(const float4*)(x + i);
  float4 b = *(const float4*)(x + i + 4);
  v8h o;
  o[0] = (_Float16)a.x; o[1] = (_Float16)a.y;
  o[2] = (_Float16)a.z; o[3] = (_Float16)a.w;
  o[4] = (_Float16)b.x; o[5] = (_Float16)b.y;
  o[6] = (_Float16)b.z; o[7] = (_Float16)b.w;
  *(v8h*)(Xh + i) = o;
}

// Fold gamma (and softmax scale for Wq) into the weight rows:
//   W'[n][k] = g[n] * W[n][k] * (p==0 ? 0.125 : 1)
__global__ __launch_bounds__(256) void cvt_w(
    const float* __restrict__ Wq, const float* __restrict__ Wk,
    const float* __restrict__ Wv, const float* __restrict__ Wo,
    const float* __restrict__ gq, const float* __restrict__ gk,
    const float* __restrict__ gv, const float* __restrict__ go,
    _Float16* __restrict__ WhAll) {
  const int p = blockIdx.z;
  const float* W = (p == 0) ? Wq : (p == 1) ? Wk : (p == 2) ? Wv : Wo;
  const float* g = (p == 0) ? gq : (p == 1) ? gk : (p == 2) ? gv : go;
  const float sc = (p == 0) ? 0.125f : 1.0f;  // SCALE = Dh^-0.5
  long i = ((long)blockIdx.x * 256 + threadIdx.x) * 8;  // within 512*512
  int n = (int)(i >> 9);
  float gs = g[n] * sc;
  float4 a = *(const float4*)(W + i);
  float4 b = *(const float4*)(W + i + 4);
  v8h o;
  o[0] = (_Float16)(a.x * gs); o[1] = (_Float16)(a.y * gs);
  o[2] = (_Float16)(a.z * gs); o[3] = (_Float16)(a.w * gs);
  o[4] = (_Float16)(b.x * gs); o[5] = (_Float16)(b.y * gs);
  o[6] = (_Float16)(b.z * gs); o[7] = (_Float16)(b.w * gs);
  *(v8h*)(WhAll + (long)p * 512 * 512 + i) = o;
}

// ---------------------------------------------------------------------------
// Q/K/V projections (pure f16 GEMM, gammas already folded).
//   mode 0: Q -> [B,H,N,Dh]   mode 1: K -> [B,H,N,Dh]   mode 2: V -> [B,H,Dh,N]
// ---------------------------------------------------------------------------
__global__ __launch_bounds__(256) void qkv_proj(
    const _Float16* __restrict__ Xh, const _Float16* __restrict__ WhAll,
    _Float16* __restrict__ Qh, _Float16* __restrict__ Kh,
    _Float16* __restrict__ Vt) {
  const int lane = threadIdx.x & 31, wave = threadIdx.x >> 5;
  const int l16 = lane & 15, half = lane >> 4;
  const int tile = blockIdx.x * 8 + wave;  // 4096 tiles = 256 (M) x 16 (N)
  const int m0 = (tile >> 4) * 32;
  const int n0 = (tile & 15) * 32;
  const int mode = blockIdx.z;
  const _Float16* Wp = WhAll + (long)mode * 512 * 512;

  v8f c00 = {}, c01 = {}, c10 = {}, c11 = {};
#pragma unroll 4
  for (int k0 = 0; k0 < 512; k0 += 32) {
    v16h a0 = load_a_h(Xh, 512, m0 + l16, k0, half);
    v16h a1 = load_a_h(Xh, 512, m0 + 16 + l16, k0, half);
    v16h b0 = load_b_h(Wp, 512, n0 + l16, k0, half);
    v16h b1 = load_b_h(Wp, 512, n0 + 16 + l16, k0, half);
    c00 = wmma_f16(a0, b0, c00);
    c01 = wmma_f16(a0, b1, c01);
    c10 = wmma_f16(a1, b0, c10);
    c11 = wmma_f16(a1, b1, c11);
  }

  auto emit = [&](v8f c, int mi, int ni) {
#pragma unroll
    for (int r = 0; r < 8; ++r) {
      int m = m0 + mi * 16 + r + 8 * half;  // row over B*N
      int n = n0 + ni * 16 + l16;           // channel 0..511
      int b = m >> 12, tok = m & 4095;
      int h = n >> 6, dh = n & 63;
      _Float16 val = (_Float16)c[r];
      if (mode == 0) {
        Qh[(((long)(b * 8 + h)) * 4096 + tok) * 64 + dh] = val;
      } else if (mode == 1) {
        Kh[(((long)(b * 8 + h)) * 4096 + tok) * 64 + dh] = val;
      } else {
        Vt[(((long)(b * 8 + h)) * 64 + dh) * 4096 + tok] = val;
      }
    }
  };
  emit(c00, 0, 0); emit(c01, 0, 1); emit(c10, 1, 0); emit(c11, 1, 1);
}

// ---------------------------------------------------------------------------
// Flash attention.  Block = 8 waves, all on the same (b,h), consecutive
// 32-row Q blocks.  K/V tiles for each 32-token KV chunk are staged into LDS
// once per block with double-buffered global_load_async_to_lds_b128; each
// wave then reads its WMMA fragments from LDS (ds_load_b128).
// ---------------------------------------------------------------------------
__global__ __launch_bounds__(256) void flash_attn(
    const _Float16* __restrict__ Qh, const _Float16* __restrict__ Kh,
    const _Float16* __restrict__ Vt, _Float16* __restrict__ Rh) {
  __shared__ _Float16 Kl[2][32 * 64];       // 2 x 4KB: K tile, [kv][dh]
  __shared__ _Float16 Vl[2][64 * 32];       // 2 x 4KB: V^T tile, [dh][kv]
  __shared__ _Float16 plds[8][2][16 * 32];  // P remap, 2KB/wave

  const int tid = threadIdx.x;
  const int lane = tid & 31, wave = tid >> 5;
  const int l16 = lane & 15, half = lane >> 4;
  const int task = blockIdx.x * 8 + wave;  // 2048 = 16 (b*h) * 128 q-blk-pairs
  const int bh = task >> 7;
  const int q0 = (task & 127) * 32;
  const _Float16* Qp = Qh + (long)bh * 4096 * 64;
  const _Float16* Kp = Kh + (long)bh * 4096 * 64;
  const _Float16* Vp = Vt + (long)bh * 64 * 4096;

  const char* KpB = (const char*)Kp;
  const char* VpB = (const char*)Vp;
  // Generic shared pointers keep the LDS byte offset in addr[31:0].
  const unsigned KlB = (unsigned)(uintptr_t)&Kl[0][0];
  const unsigned VlB = (unsigned)(uintptr_t)&Vl[0][0];

  // Stage one 32-token KV chunk: 256 threads x 16B cover each 4KB tile.
  auto stage = [&](int buf, int kv) {
    unsigned kl = KlB + buf * 4096 + tid * 16;
    unsigned long long kg =
        (unsigned long long)(KpB + (long)kv * 128 + tid * 16);
    asm volatile("global_load_async_to_lds_b128 %0, %1, off" ::"v"(kl),
                 "v"(kg)
                 : "memory");
    int seg = tid >> 2, off = (tid & 3) * 16;  // V: 64 rows x 64B, stride 8KB
    unsigned vl = VlB + buf * 4096 + seg * 64 + off;
    unsigned long long vg =
        (unsigned long long)(VpB + (long)seg * 8192 + (long)kv * 2 + off);
    asm volatile("global_load_async_to_lds_b128 %0, %1, off" ::"v"(vl),
                 "v"(vg)
                 : "memory");
  };

  v16h qa[2][2];  // [q sub-block][dh chunk]
#pragma unroll
  for (int qb = 0; qb < 2; ++qb) {
    qa[qb][0] = load_a_h(Qp, 64, q0 + qb * 16 + l16, 0, half);
    qa[qb][1] = load_a_h(Qp, 64, q0 + qb * 16 + l16, 32, half);
  }

  v8f o[2][4];
  float mrun[2][8], lrun[2][8];
#pragma unroll
  for (int qb = 0; qb < 2; ++qb) {
#pragma unroll
    for (int d = 0; d < 4; ++d) o[qb][d] = (v8f){};
#pragma unroll
    for (int r = 0; r < 8; ++r) { mrun[qb][r] = -1e30f; lrun[qb][r] = 0.f; }
  }

  stage(0, 0);  // prime the pipeline

  for (int t = 0; t < 128; ++t) {
    const int cur = t & 1;
    if (t + 1 < 128) {
      stage(cur ^ 1, (t + 1) * 32);
      // two older async loads (chunk t) complete in-order
      asm volatile("s_wait_asynccnt 2" ::: "memory");
    } else {
      asm volatile("s_wait_asynccnt 0" ::: "memory");
    }
    __syncthreads();  // chunk t visible to all waves
    const _Float16* Kc = &Kl[cur][0];
    const _Float16* Vc = &Vl[cur][0];

    // ---- S = (Q*scale) @ K^T : 2 q-blocks x 2 kv-tiles, k=64 over Dh ----
    v8f s[2][2] = {{(v8f){}, (v8f){}}, {(v8f){}, (v8f){}}};
#pragma unroll
    for (int kt = 0; kt < 2; ++kt) {
#pragma unroll
      for (int dc = 0; dc < 2; ++dc) {
        v16h kb = load_b_h(Kc, 64, 16 * kt + l16, 32 * dc, half);
        s[0][kt] = wmma_f16(qa[0][dc], kb, s[0][kt]);
        s[1][kt] = wmma_f16(qa[1][dc], kb, s[1][kt]);
      }
    }
    // ---- online softmax per q-block; xor masks 1/2/4/8 stay in a half ----
#pragma unroll
    for (int qb = 0; qb < 2; ++qb) {
      _Float16* lp = &plds[wave][qb][0];
#pragma unroll
      for (int r = 0; r < 8; ++r) {
        float mx = fmaxf(s[qb][0][r], s[qb][1][r]);
        mx = fmaxf(mx, __shfl_xor(mx, 1));
        mx = fmaxf(mx, __shfl_xor(mx, 2));
        mx = fmaxf(mx, __shfl_xor(mx, 4));
        mx = fmaxf(mx, __shfl_xor(mx, 8));
        float mnew = fmaxf(mrun[qb][r], mx);
        float alpha = __expf(mrun[qb][r] - mnew);
        float p0 = __expf(s[qb][0][r] - mnew);
        float p1 = __expf(s[qb][1][r] - mnew);
        float ps = p0 + p1;
        ps += __shfl_xor(ps, 1);
        ps += __shfl_xor(ps, 2);
        ps += __shfl_xor(ps, 4);
        ps += __shfl_xor(ps, 8);
        lrun[qb][r] = lrun[qb][r] * alpha + ps;
        mrun[qb][r] = mnew;
        o[qb][0][r] *= alpha; o[qb][1][r] *= alpha;
        o[qb][2][r] *= alpha; o[qb][3][r] *= alpha;
        lp[(r + 8 * half) * 32 + l16] = (_Float16)p0;
        lp[(r + 8 * half) * 32 + 16 + l16] = (_Float16)p1;
      }
    }
    // ---- re-read P tiles as A-fragments (same-wave LDS is in-order) ----
    v16h pa[2];
#pragma unroll
    for (int qb = 0; qb < 2; ++qb) {
      const _Float16* q = &plds[wave][qb][0] + l16 * 32 + 8 * half;
      v8h lo = *(const v8h*)q;
      v8h hi = *(const v8h*)(q + 16);
      pa[qb] = cat8(lo, hi);
    }
    // ---- O += P @ V ; each V fragment (from LDS) feeds both q-blocks ----
#pragma unroll
    for (int d = 0; d < 4; ++d) {
      v16h vb = load_b_h(Vc, 32, d * 16 + l16, 0, half);
      o[0][d] = wmma_f16(pa[0], vb, o[0][d]);
      o[1][d] = wmma_f16(pa[1], vb, o[1][d]);
    }
    __syncthreads();  // all waves done with buf[cur] before its reuse
  }

  // ---- r = O / l, merged-head layout [B, N, 512] in f16 ----
  const int b = bh >> 3, h = bh & 7;
#pragma unroll
  for (int qb = 0; qb < 2; ++qb) {
#pragma unroll
    for (int d = 0; d < 4; ++d) {
#pragma unroll
      for (int r = 0; r < 8; ++r) {
        int tok = q0 + qb * 16 + r + 8 * half;
        int n = h * 64 + d * 16 + l16;
        Rh[((long)b * 4096 + tok) * 512 + n] =
            (_Float16)(o[qb][d][r] / lrun[qb][r]);
      }
    }
  }
}

// ---------------------------------------------------------------------------
// out = r @ Wo'^T + gamma_out*bo   (gamma_out folded into Wo'), f32 output.
// ---------------------------------------------------------------------------
__global__ __launch_bounds__(256) void out_proj(
    const _Float16* __restrict__ Rh, const _Float16* __restrict__ WoH,
    const float* __restrict__ bo, const float* __restrict__ gout,
    float* __restrict__ out) {
  const int lane = threadIdx.x & 31, wave = threadIdx.x >> 5;
  const int l16 = lane & 15, half = lane >> 4;
  const int tile = blockIdx.x * 8 + wave;
  const int m0 = (tile >> 4) * 32;
  const int n0 = (tile & 15) * 32;

  v8f c00 = {}, c01 = {}, c10 = {}, c11 = {};
#pragma unroll 4
  for (int k0 = 0; k0 < 512; k0 += 32) {
    v16h a0 = load_a_h(Rh, 512, m0 + l16, k0, half);
    v16h a1 = load_a_h(Rh, 512, m0 + 16 + l16, k0, half);
    v16h b0 = load_b_h(WoH, 512, n0 + l16, k0, half);
    v16h b1 = load_b_h(WoH, 512, n0 + 16 + l16, k0, half);
    c00 = wmma_f16(a0, b0, c00);
    c01 = wmma_f16(a0, b1, c01);
    c10 = wmma_f16(a1, b0, c10);
    c11 = wmma_f16(a1, b1, c11);
  }
  auto emit = [&](v8f c, int mi, int ni) {
#pragma unroll
    for (int r = 0; r < 8; ++r) {
      int m = m0 + mi * 16 + r + 8 * half;
      int n = n0 + ni * 16 + l16;
      out[(long)m * 512 + n] = c[r] + gout[n] * bo[n];
    }
  };
  emit(c00, 0, 0); emit(c01, 0, 1); emit(c10, 1, 0); emit(c11, 1, 1);
}

// ---------------------------------------------------------------------------
extern "C" void kernel_launch(void* const* d_in, const int* in_sizes, int n_in,
                              void* d_out, int out_size, void* d_ws,
                              size_t ws_size, hipStream_t stream) {
  (void)in_sizes; (void)n_in; (void)out_size; (void)ws_size;
  const float* x = (const float*)d_in[0];
  const float* Wq = (const float*)d_in[1];
  const float* Wk = (const float*)d_in[2];
  const float* Wv = (const float*)d_in[3];
  const float* Wo = (const float*)d_in[4];
  const float* bo = (const float*)d_in[5];
  const float* gq = (const float*)d_in[6];
  const float* gk = (const float*)d_in[7];
  const float* gv = (const float*)d_in[8];
  const float* go = (const float*)d_in[9];
  float* out = (float*)d_out;

  const long E = 2L * 8 * 4096 * 64;  // 4,194,304 f16 elems (8 MB) each
  _Float16* Qh = (_Float16*)d_ws;
  _Float16* Kh = Qh + E;
  _Float16* Vt = Kh + E;
  _Float16* Rh = Vt + E;
  _Float16* Xh = Rh + E;            // 8192x512 f16
  _Float16* Wh = Xh + E;            // 4 x 512x512 f16 (Wq',Wk',Wv',Wo')

  cvt_x<<<2048, 256, 0, stream>>>(x, Xh);
  cvt_w<<<dim3(128, 1, 4), 256, 0, stream>>>(Wq, Wk, Wv, Wo, gq, gk, gv, go, Wh);
  qkv_proj<<<dim3(512, 1, 3), 256, 0, stream>>>(Xh, Wh, Qh, Kh, Vt);
  flash_attn<<<256, 256, 0, stream>>>(Qh, Kh, Vt, Rh);
  out_proj<<<512, 256, 0, stream>>>(Rh, Wh + 3L * 512 * 512, bo, go, out);
}
